// MultiHeadedSelfAttention_8272107012563
// MI455X (gfx1250) — matrix operand order repaired and checked by reading, not compiled, not run on hardware
//
#include <hip/hip_runtime.h>
#include <hip/hip_bf16.h>

typedef __attribute__((ext_vector_type(16))) _Float16 v16h;
typedef __attribute__((ext_vector_type(8)))  _Float16 h8;
typedef __attribute__((ext_vector_type(8)))  float    v8f;

union F16Frag { v16h v; h8 h[2]; };

#define B_   8
#define S_   1024
#define D_   768
#define H_   12
#define W_   64
#define BS_  8192
#define DD_  589824      // D*D
#define BSD_ 6291456     // B*S*D

__device__ __forceinline__ float fq_apply(float x, float scale, float invscale) {
    return fminf(fmaxf(rintf(x * scale), -32768.0f), 32767.0f) * invscale;
}
__device__ __forceinline__ float fq_scale(float maxabs) {
    float mb = 15.0f - ceilf(log2f(fmaxf(maxabs, 1e-8f)));
    return exp2f(mb);
}

__device__ __forceinline__ v8f wmma_f16(v16h a, v16h b, v8f c) {
    return __builtin_amdgcn_wmma_f32_16x16x32_f16(false, a, false, b, (short)0, c, false, false);
}

// gfx1250 async copy: 16B per lane, global -> LDS, tracked by ASYNCcnt
__device__ __forceinline__ void async_copy_b128(_Float16* ldst, const _Float16* gsrc) {
    unsigned int loff = (unsigned int)(uintptr_t)ldst;     // LDS aperture: low 32 bits = LDS address
    unsigned long long gaddr = (unsigned long long)(uintptr_t)gsrc;
    asm volatile("global_load_async_to_lds_b128 %0, %1, off"
                 :: "v"(loff), "v"(gaddr) : "memory");
}
__device__ __forceinline__ void wait_async0() {
    asm volatile("s_wait_asynccnt 0x0" ::: "memory");
}

// A-matrix fragment (16x32 f16): lanes 0-15 row m hold K {0..15}, lanes 16-31 hold K {16..31}
__device__ __forceinline__ v16h load_afrag(const _Float16* __restrict__ base, int ld, int m0, int kk, int lane) {
    int row = m0 + (lane & 15);
    int off = (lane & 16) ? 16 : 0;
    const _Float16* p = base + (size_t)row * ld + kk + off;
    F16Frag f;
    f.h[0] = *(const h8*)(p);
    f.h[1] = *(const h8*)(p + 8);
    return f.v;
}
// B-matrix fragment (32x16 f16) loaded from B^T stored row-major (rows=N, cols=K)
__device__ __forceinline__ v16h load_bfrag(const _Float16* __restrict__ baseT, int ld, int n0, int kk, int lane) {
    int row = n0 + (lane & 15);
    int off = (lane & 16) ? 16 : 0;
    const _Float16* p = baseT + (size_t)row * ld + kk + off;
    F16Frag f;
    f.h[0] = *(const h8*)(p);
    f.h[1] = *(const h8*)(p + 8);
    return f.v;
}

__device__ __forceinline__ float wave_max(float v) {
#pragma unroll
    for (int o = 16; o > 0; o >>= 1) v = fmaxf(v, __shfl_xor(v, o, 32));
    return v;
}
__device__ __forceinline__ float wave_min(float v) {
#pragma unroll
    for (int o = 16; o > 0; o >>= 1) v = fminf(v, __shfl_xor(v, o, 32));
    return v;
}

// stats[0..2] = max|q|,|k|,|v| raw; stats[3] = max|scores|; stats[4] = min Z; stats[5] = max|h|
__global__ void k_init_stats(float* stats) {
    if (threadIdx.x == 0) {
        stats[0] = 0.f; stats[1] = 0.f; stats[2] = 0.f;
        stats[3] = 0.f; stats[4] = 3.4e38f; stats[5] = 0.f;
    }
}

__global__ void k_pack(const float* __restrict__ x,
                       const float* __restrict__ Wq, const float* __restrict__ Wk, const float* __restrict__ Wv,
                       _Float16* __restrict__ xh, _Float16* __restrict__ wh) {
    size_t stride = (size_t)gridDim.x * blockDim.x;
    size_t tid = (size_t)blockIdx.x * blockDim.x + threadIdx.x;
    for (size_t i = tid; i < (size_t)BSD_; i += stride) xh[i] = (_Float16)x[i];
    for (size_t i = tid; i < (size_t)DD_; i += stride) {
        wh[i]               = (_Float16)Wq[i];
        wh[(size_t)DD_ + i] = (_Float16)Wk[i];
        wh[2*(size_t)DD_+i] = (_Float16)Wv[i];
    }
}

// y = x @ W^T + b ; raw fp32 out + per-tensor max|y|
// Weight tile (shared by all 8 waves) staged in LDS via async-to-LDS, double buffered.
__global__ __launch_bounds__(256) void k_gemm_qkv(
        const _Float16* __restrict__ xh, const _Float16* __restrict__ wh,
        const float* __restrict__ bq, const float* __restrict__ bk, const float* __restrict__ bv,
        float* __restrict__ raw, float* __restrict__ stats) {
    __shared__ __align__(16) _Float16 wtile[2][64][32];   // 2 x 4KB: B^T rows (n) x K-chunk
    int tid = threadIdx.x;
    int lane = tid & 31, wid = tid >> 5;
    int m0 = blockIdx.x * 128 + wid * 16;
    int n0 = blockIdx.y * 64;
    int wsel = blockIdx.z;
    const _Float16* Wt = wh + (size_t)wsel * DD_;       // B^T of (x@W^T) is W itself
    const float* bias = (wsel == 0) ? bq : ((wsel == 1) ? bk : bv);

    // per-thread copy slot: 256 threads cover 64 rows x 32 halves (16B each)
    int crow = tid >> 2;            // 0..63
    int cseg = (tid & 3) << 3;      // half offset 0,8,16,24
    const _Float16* gsrc = Wt + (size_t)(n0 + crow) * D_ + cseg;

    async_copy_b128(&wtile[0][crow][cseg], gsrc);       // stage kk = 0

    v8f acc[4] = {};
    for (int kk = 0; kk < D_; kk += 32) {
        int cur = (kk >> 5) & 1;
        wait_async0();              // own async stores for wtile[cur] complete
        __syncthreads();            // everyone's copies visible; prior reads of wtile[cur^1] done
        if (kk + 32 < D_) async_copy_b128(&wtile[cur ^ 1][crow][cseg], gsrc + kk + 32);
        v16h a = load_afrag(xh, D_, m0, kk, lane);
#pragma unroll
        for (int j = 0; j < 4; ++j) {
            int row = 16 * j + (lane & 15);
            int off = (lane & 16) ? 16 : 0;
            const _Float16* p = &wtile[cur][row][off];
            F16Frag f;
            f.h[0] = *(const h8*)p;
            f.h[1] = *(const h8*)(p + 8);
            acc[j] = wmma_f16(a, f.v, acc[j]);
        }
    }
    int moff = (lane >> 4) << 3;
    float mx = 0.0f;
#pragma unroll
    for (int j = 0; j < 4; ++j) {
        int n = n0 + 16 * j + (lane & 15);
        float bn = bias[n];
#pragma unroll
        for (int i = 0; i < 8; ++i) {
            int m = m0 + i + moff;
            float val = acc[j][i] + bn;
            raw[((size_t)wsel * BS_ + m) * D_ + n] = val;
            mx = fmaxf(mx, fabsf(val));
        }
    }
    mx = wave_max(mx);
    if (lane == 0) atomicMax((unsigned int*)&stats[wsel], __float_as_uint(mx));
}

__global__ void k_quant_qkv(const float* __restrict__ raw, const float* __restrict__ stats,
                            _Float16* __restrict__ qh, _Float16* __restrict__ kh,
                            _Float16* __restrict__ vT) {
    float sq = fq_scale(stats[0]), sk = fq_scale(stats[1]), sv = fq_scale(stats[2]);
    float iq = 1.0f / sq, ik = 1.0f / sk, iv = 1.0f / sv;
    size_t stride = (size_t)gridDim.x * blockDim.x;
    for (size_t i = (size_t)blockIdx.x * blockDim.x + threadIdx.x; i < (size_t)BSD_; i += stride) {
        qh[i] = (_Float16)fq_apply(raw[i], sq, iq);
        kh[i] = (_Float16)fq_apply(raw[(size_t)BSD_ + i], sk, ik);
        float vq = fq_apply(raw[2 * (size_t)BSD_ + i], sv, iv);
        int d = (int)(i % D_);
        int bs = (int)(i / D_);
        int b = bs / S_, s = bs % S_;
        int hh = d / W_, w = d % W_;
        vT[(((size_t)b * H_ + hh) * W_ + w) * S_ + s] = (_Float16)vq;   // v^T per (b,h): [W][S]
    }
}

// pass 1 over scores: only max|scores| (scores never materialized)
__global__ __launch_bounds__(256) void k_scores_max(
        const _Float16* __restrict__ qh, const _Float16* __restrict__ kh, float* __restrict__ stats) {
    int lane = threadIdx.x & 31, wid = threadIdx.x >> 5;
    int m0 = blockIdx.x * 128 + wid * 16;
    int t0 = blockIdx.y * 64;
    int bh = blockIdx.z, b = bh / H_, h = bh % H_;
    const _Float16* qb = qh + (size_t)b * S_ * D_ + h * W_;
    const _Float16* kb = kh + (size_t)b * S_ * D_ + h * W_;
    v8f acc[4] = {};
#pragma unroll
    for (int k2 = 0; k2 < 2; ++k2) {
        v16h a = load_afrag(qb, D_, m0, 32 * k2, lane);
#pragma unroll
        for (int j = 0; j < 4; ++j) {
            v16h bb = load_bfrag(kb, D_, t0 + 16 * j, 32 * k2, lane);
            acc[j] = wmma_f16(a, bb, acc[j]);
        }
    }
    float mx = 0.0f;
#pragma unroll
    for (int j = 0; j < 4; ++j)
#pragma unroll
        for (int i = 0; i < 8; ++i) mx = fmaxf(mx, fabsf(acc[j][i]));
    mx = wave_max(mx);
    if (lane == 0) atomicMax((unsigned int*)&stats[3], __float_as_uint(mx));
}

// pass 2: per-row softmax stats (m, 1/Z) of quantized+scaled+masked scores; global min Z
__global__ __launch_bounds__(256) void k_rowstats(
        const _Float16* __restrict__ qh, const _Float16* __restrict__ kh,
        const float* __restrict__ mask, float* __restrict__ stats,
        float2* __restrict__ rowst) {
    __shared__ __align__(16) float tile[8][16][64];
    int lane = threadIdx.x & 31, wid = threadIdx.x >> 5;
    int m0 = blockIdx.x * 128 + wid * 16;
    int bh = blockIdx.y, b = bh / H_, h = bh % H_;
    const _Float16* qb = qh + (size_t)b * S_ * D_ + h * W_;
    const _Float16* kb = kh + (size_t)b * S_ * D_ + h * W_;
    const float* mb = mask + (size_t)b * S_;
    float scale1 = fq_scale(stats[3]);            float inv1 = 1.0f / scale1;
    float qm     = fq_apply(stats[3], scale1, inv1);
    float scale2 = fq_scale(qm * 0.125f);         float inv2 = 1.0f / scale2;
    v16h a0 = load_afrag(qb, D_, m0, 0, lane);
    v16h a1 = load_afrag(qb, D_, m0, 32, lane);
    int moff = (lane >> 4) << 3;
    int lr = lane & 15, hi = lane >> 4;
    float mrun = -3.4e38f, zrun = 0.0f;
    for (int t0 = 0; t0 < S_; t0 += 64) {
        v8f acc[4] = {};
#pragma unroll
        for (int j = 0; j < 4; ++j) {
            acc[j] = wmma_f16(a0, load_bfrag(kb, D_, t0 + 16 * j, 0, lane), acc[j]);
            acc[j] = wmma_f16(a1, load_bfrag(kb, D_, t0 + 16 * j, 32, lane), acc[j]);
        }
#pragma unroll
        for (int j = 0; j < 4; ++j) {
            float madd = -10000.0f * (1.0f - mb[t0 + 16 * j + (lane & 15)]);
#pragma unroll
            for (int i = 0; i < 8; ++i) {
                float s1v = fq_apply(acc[j][i], scale1, inv1);
                float s2v = fq_apply(s1v * 0.125f, scale2, inv2);
                tile[wid][i + moff][16 * j + (lane & 15)] = s2v + madd;
            }
        }
        float cm = -3.4e38f;
#pragma unroll
        for (int e = 0; e < 32; ++e) cm = fmaxf(cm, tile[wid][lr][hi * 32 + e]);
        cm = fmaxf(cm, __shfl_xor(cm, 16, 32));
        float cs = 0.0f;
#pragma unroll
        for (int e = 0; e < 32; ++e) cs += __expf(tile[wid][lr][hi * 32 + e] - cm);
        cs += __shfl_xor(cs, 16, 32);
        if (cm > mrun) { zrun = zrun * __expf(mrun - cm) + cs; mrun = cm; }
        else           { zrun += cs * __expf(cm - mrun); }
    }
    if (hi == 0) {
        float2 rs; rs.x = mrun; rs.y = 1.0f / zrun;
        rowst[(size_t)bh * S_ + m0 + lr] = rs;
    }
    float zmin = wave_min(zrun);
    if (lane == 0) atomicMin((unsigned int*)&stats[4], __float_as_uint(zmin));
}

// pass 3: recompute scores, quantize -> softmax -> quantize probs -> LDS reshape -> P @ v
__global__ __launch_bounds__(256) void k_attn(
        const _Float16* __restrict__ qh, const _Float16* __restrict__ kh,
        const _Float16* __restrict__ vT, const float* __restrict__ mask,
        float* __restrict__ stats, const float2* __restrict__ rowst,
        float* __restrict__ out) {
    __shared__ __align__(16) _Float16 ptile[8][16][32];
    int lane = threadIdx.x & 31, wid = threadIdx.x >> 5;
    int m0 = blockIdx.x * 128 + wid * 16;
    int bh = blockIdx.y, b = bh / H_, h = bh % H_;
    const _Float16* qb = qh + (size_t)b * S_ * D_ + h * W_;
    const _Float16* kb = kh + (size_t)b * S_ * D_ + h * W_;
    const _Float16* vb = vT + (size_t)bh * W_ * S_;
    const float* mb = mask + (size_t)b * S_;
    float scale1 = fq_scale(stats[3]);            float inv1 = 1.0f / scale1;
    float qm     = fq_apply(stats[3], scale1, inv1);
    float scale2 = fq_scale(qm * 0.125f);         float inv2 = 1.0f / scale2;
    float maxp   = 1.0f / stats[4];               // max prob = 1 / min Z
    float scalep = fq_scale(maxp);                float invp = 1.0f / scalep;
    v16h a0 = load_afrag(qb, D_, m0, 0, lane);
    v16h a1 = load_afrag(qb, D_, m0, 32, lane);
    int moff = (lane >> 4) << 3;
    float rm[8], rz[8];
#pragma unroll
    for (int i = 0; i < 8; ++i) {
        float2 rs = rowst[(size_t)bh * S_ + m0 + i + moff];
        rm[i] = rs.x; rz[i] = rs.y;
    }
    v8f hacc[4] = {};
    for (int t0 = 0; t0 < S_; t0 += 32) {
        if (t0 + 32 < S_) {   // latency hint for next K rows -> global_prefetch_b8
            __builtin_prefetch(kb + (size_t)(t0 + 32 + (lane & 15)) * D_, 0, 1);
            __builtin_prefetch(kb + (size_t)(t0 + 48 + (lane & 15)) * D_, 0, 1);
        }
        v8f s0 = {}, s1 = {};
        s0 = wmma_f16(a0, load_bfrag(kb, D_, t0, 0, lane), s0);
        s0 = wmma_f16(a1, load_bfrag(kb, D_, t0, 32, lane), s0);
        s1 = wmma_f16(a0, load_bfrag(kb, D_, t0 + 16, 0, lane), s1);
        s1 = wmma_f16(a1, load_bfrag(kb, D_, t0 + 16, 32, lane), s1);
        float madd0 = -10000.0f * (1.0f - mb[t0 + (lane & 15)]);
        float madd1 = -10000.0f * (1.0f - mb[t0 + 16 + (lane & 15)]);
#pragma unroll
        for (int i = 0; i < 8; ++i) {
            float sa = fq_apply(s0[i], scale1, inv1);
            sa = fq_apply(sa * 0.125f, scale2, inv2) + madd0;
            float pa = fq_apply(__expf(sa - rm[i]) * rz[i], scalep, invp);
            ptile[wid][i + moff][lane & 15] = (_Float16)pa;
            float sb = fq_apply(s1[i], scale1, inv1);
            sb = fq_apply(sb * 0.125f, scale2, inv2) + madd1;
            float pb = fq_apply(__expf(sb - rm[i]) * rz[i], scalep, invp);
            ptile[wid][i + moff][16 + (lane & 15)] = (_Float16)pb;
        }
        // reshape probs C-tile (in LDS) into an A-fragment (intra-wave, DS in-order)
        int arow = lane & 15;
        int aoff = (lane & 16) ? 16 : 0;
        F16Frag pf;
        pf.h[0] = *(const h8*)&ptile[wid][arow][aoff];
        pf.h[1] = *(const h8*)&ptile[wid][arow][aoff + 8];
#pragma unroll
        for (int j = 0; j < 4; ++j) {
            v16h vv = load_bfrag(vb, S_, 16 * j, t0, lane);   // B^T of (P@v) is v^T, rows = w
            hacc[j] = wmma_f16(pf.v, vv, hacc[j]);
        }
    }
    float mx = 0.0f;
#pragma unroll
    for (int j = 0; j < 4; ++j) {
        int w = 16 * j + (lane & 15);
#pragma unroll
        for (int i = 0; i < 8; ++i) {
            int m = m0 + i + moff;
            float val = hacc[j][i];
            out[((size_t)b * S_ + m) * D_ + h * W_ + w] = val;
            mx = fmaxf(mx, fabsf(val));
        }
    }
    mx = wave_max(mx);
    if (lane == 0) atomicMax((unsigned int*)&stats[5], __float_as_uint(mx));
}

__global__ void k_quant_out(float* __restrict__ out, const float* __restrict__ stats) {
    float s = fq_scale(stats[5]);
    float is = 1.0f / s;
    size_t stride = (size_t)gridDim.x * blockDim.x;
    for (size_t i = (size_t)blockIdx.x * blockDim.x + threadIdx.x; i < (size_t)BSD_; i += stride)
        out[i] = fq_apply(out[i], s, is);
}

extern "C" void kernel_launch(void* const* d_in, const int* in_sizes, int n_in,
                              void* d_out, int out_size, void* d_ws, size_t ws_size,
                              hipStream_t stream) {
    (void)in_sizes; (void)n_in; (void)out_size; (void)ws_size;
    const float* x    = (const float*)d_in[0];
    const float* mask = (const float*)d_in[1];
    const float* Wq   = (const float*)d_in[2];
    const float* bq   = (const float*)d_in[3];
    const float* Wk   = (const float*)d_in[4];
    const float* bk   = (const float*)d_in[5];
    const float* Wv   = (const float*)d_in[6];
    const float* bv   = (const float*)d_in[7];
    float* out = (float*)d_out;

    char* ws = (char*)d_ws;
    size_t o = 0;
    auto carve = [&](size_t bytes) { size_t p = o; o += (bytes + 255) & ~(size_t)255; return p; };
    float*    stats = (float*)   (ws + carve(64 * sizeof(float)));
    float2*   rstat = (float2*)  (ws + carve((size_t)B_ * H_ * S_ * sizeof(float2)));
    _Float16* xh    = (_Float16*)(ws + carve((size_t)BSD_ * 2));
    _Float16* wh    = (_Float16*)(ws + carve((size_t)3 * DD_ * 2));
    float*    raw   = (float*)   (ws + carve((size_t)3 * BSD_ * 4));
    _Float16* qqh   = (_Float16*)(ws + carve((size_t)BSD_ * 2));
    _Float16* kkh   = (_Float16*)(ws + carve((size_t)BSD_ * 2));
    _Float16* vT    = (_Float16*)(ws + carve((size_t)BSD_ * 2));

    k_init_stats<<<1, 32, 0, stream>>>(stats);
    k_pack<<<2048, 256, 0, stream>>>(x, Wq, Wk, Wv, xh, wh);
    dim3 g1(BS_ / 128, D_ / 64, 3);
    k_gemm_qkv<<<g1, 256, 0, stream>>>(xh, wh, bq, bk, bv, raw, stats);
    k_quant_qkv<<<2048, 256, 0, stream>>>(raw, stats, qqh, kkh, vT);
    dim3 g2(S_ / 128, S_ / 64, B_ * H_);
    k_scores_max<<<g2, 256, 0, stream>>>(qqh, kkh, stats);
    dim3 g3(S_ / 128, B_ * H_);
    k_rowstats<<<g3, 256, 0, stream>>>(qqh, kkh, mask, stats, rstat);
    k_attn<<<g3, 256, 0, stream>>>(qqh, kkh, vT, mask, stats, rstat, out);
    k_quant_out<<<2048, 256, 0, stream>>>(out, stats);
}
